// Edge_node_50869592655538
// MI455X (gfx1250) — matrix operand order repaired and checked by reading, not compiled, not run on hardware
//
#include <hip/hip_runtime.h>

// ---------------------------------------------------------------------------
// Edge/Node GNN block for MI455X (gfx1250, wave32).
// bf16 WMMA (V_WMMA_F32_16X16X32_BF16) with f32 accumulation; BN stats fused
// into GEMM epilogues; segment sums via native f32 global atomics.
// Wave tile: 16x128 (8 WMMAs per A fragment), block 8 waves = 128x128.
// ---------------------------------------------------------------------------

typedef __bf16 bf16;
typedef __attribute__((ext_vector_type(16))) __bf16 v16bf;
typedef __attribute__((ext_vector_type(8)))  float  v8f;

#define HDIM 128
#define BN_EPS 1e-5f

__device__ __forceinline__ void atomAddF(float* p, float v) {
  unsafeAtomicAdd(p, v);   // lowers to global_atomic_add_f32 on gfx1250
}

// ---------------------------------------------------------------------------
// Pack f32 weight [K,Nc] (row-major) into WMMA bf16 B-fragment order:
// pk[((kt*NT+nt)*32+lane)*16+j] = W[kt*32 + 16*(lane>>4) + j][nt*16 + (lane&15)]
// so each lane's 16 bf16 are contiguous (two b128 loads per fragment).
// ---------------------------------------------------------------------------
__global__ void pack_w(const float* __restrict__ W, bf16* __restrict__ out,
                       int K, int Nc) {
  int tid = blockIdx.x * 256 + threadIdx.x;
  int total = K * Nc;
  if (tid >= total) return;
  int j    = tid & 15;
  int lane = (tid >> 4) & 31;
  int rest = tid >> 9;
  int ntiles = Nc >> 4;
  int nt = rest % ntiles;
  int kt = rest / ntiles;
  int k = kt * 32 + 16 * (lane >> 4) + j;
  int n = nt * 16 + (lane & 15);
  out[tid] = (bf16)W[(size_t)k * Nc + n];
}

// ---------------------------------------------------------------------------
// GEMM: C[M,Nt] = act(A[M,K]) @ Bpk, A bf16 row-major, B fragment-packed bf16.
// Optional fused BN+ReLU on A (aScale/aShift per K-column, null = identity).
// Fused per-column stats (sum, sum-of-squares) via LDS reduce + global atomic.
// Block: 256 threads = 8 waves; each wave computes a 16x128 tile (8 WMMA acc).
// Grid: x = ceil(M/128), y = Nt/128.
// ---------------------------------------------------------------------------
template <bool OUT_BF16>
__global__ __launch_bounds__(256) void gemm_bn(
    const bf16* __restrict__ A, int M, int K,
    const bf16* __restrict__ Bpk, int Nt,
    const float* __restrict__ aScale, const float* __restrict__ aShift,
    void* __restrict__ Cout,
    float* __restrict__ stSum, float* __restrict__ stSq) {
  __shared__ __align__(16) unsigned char ldsA[128 * 80];  // 128 rows x 32 bf16, padded pitch
  __shared__ float ldsSum[128];
  __shared__ float ldsSq[128];

  const int tid  = threadIdx.x;
  const int lane = tid & 31;
  const int wv   = tid >> 5;
  const int half = lane >> 4;
  const int nloc = lane & 15;
  const int row0 = blockIdx.x * 128;
  const int colBlock = blockIdx.y * 128;
  const int ntiles = Nt >> 4;

  v8f acc[8];
#pragma unroll
  for (int t = 0; t < 8; ++t)
#pragma unroll
    for (int r = 0; r < 8; ++r) acc[t][r] = 0.f;

  const int nk = K >> 5;
  for (int kt = 0; kt < nk; ++kt) {
    // ---- stage A tile (128 rows x 32 K, bf16) into LDS, coalesced b128 ----
#pragma unroll
    for (int p = 0; p < 2; ++p) {
      int idx = p * 256 + tid;
      int r = idx >> 2, seg = idx & 3;
      int rg = row0 + r;
      uint4 q = make_uint4(0u, 0u, 0u, 0u);
      if (rg < M) {
        const bf16* ap = A + (size_t)rg * K + kt * 32 + seg * 8;
        q = *(const uint4*)ap;
        if (kt + 1 < nk)  // warm WGP$/L2 for next k-step (global_prefetch_b8)
          __builtin_prefetch(ap + 32, 0, 1);
      }
      if (aScale) {  // fused BN+ReLU of previous layer, per K-column
        union { uint4 q; bf16 b[8]; } u; u.q = q;
        int c0 = kt * 32 + seg * 8;
#pragma unroll
        for (int i = 0; i < 8; ++i) {
          float f = (float)u.b[i] * aScale[c0 + i] + aShift[c0 + i];
          u.b[i] = (bf16)fmaxf(f, 0.f);
        }
        q = u.q;
      }
      *(uint4*)(ldsA + r * 80 + seg * 16) = q;
    }
    __syncthreads();

    // ---- A fragment, contiguous K order matching packed B: K runs [16h..16h+15] ----
    const unsigned char* ar = ldsA + (wv * 16 + nloc) * 80;
    union { uint4 q[2]; v16bf v; } af;
    af.q[0] = *(const uint4*)(ar + half * 32);
    af.q[1] = *(const uint4*)(ar + half * 32 + 16);

#pragma unroll
    for (int t = 0; t < 8; ++t) {
      int nt = (colBlock >> 4) + t;
      const v16bf bv =
          *(const v16bf*)(Bpk + ((size_t)(kt * ntiles + nt) * 32 + lane) * 16);
      acc[t] = __builtin_amdgcn_wmma_f32_16x16x32_bf16(
          false, af.v, false, bv, (short)0, acc[t], false, false);
    }
    __syncthreads();
  }

  // ---- epilogue: store + per-column stats (LDS reduce, 1 atomic/col/block) ----
  if (tid < 128) { ldsSum[tid] = 0.f; ldsSq[tid] = 0.f; }
  __syncthreads();
#pragma unroll
  for (int t = 0; t < 8; ++t) {
    int col  = t * 16 + nloc;
    int gcol = colBlock + col;
    float ps = 0.f, pq = 0.f;
#pragma unroll
    for (int r = 0; r < 8; ++r) {
      int rg = row0 + wv * 16 + 8 * half + r;  // C layout: VGPR r <-> M = r + 8*half
      float v = acc[t][r];
      if (rg < M) {
        if (OUT_BF16)
          ((bf16*)Cout)[(size_t)rg * Nt + gcol] = (bf16)v;
        else
          ((float*)Cout)[(size_t)rg * Nt + gcol] = v;
        ps += v; pq += v * v;
      }
    }
    atomicAdd(&ldsSum[col], ps);  // ds_add_f32
    atomicAdd(&ldsSq[col], pq);
  }
  __syncthreads();
  if (tid < 128) {
    atomAddF(&stSum[colBlock + tid], ldsSum[tid]);
    atomAddF(&stSq[colBlock + tid], ldsSq[tid]);
  }
}

// ---------------------------------------------------------------------------
// BN finalize: sc = g * rsqrt(var+eps), sh = b - mean*sc  (biased variance)
// ---------------------------------------------------------------------------
__global__ void bn_finalize(const float* __restrict__ sum,
                            const float* __restrict__ sq,
                            const float* __restrict__ g,
                            const float* __restrict__ b, float invM, int n,
                            float* __restrict__ sc, float* __restrict__ sh) {
  int t = threadIdx.x;
  if (t < n) {
    float m = sum[t] * invM;
    float v = sq[t] * invM - m * m;
    float a = g[t] * rsqrtf(v + BN_EPS);
    sc[t] = a;
    sh[t] = b[t] - m * a;
  }
}

// ---------------------------------------------------------------------------
// S[n] = segsum(X by src) + segsum(X by dst); one thread per (edge, 4 cols)
// ---------------------------------------------------------------------------
__global__ void scatter_sum2(const float* __restrict__ X,
                             const int* __restrict__ src,
                             const int* __restrict__ dst,
                             float* __restrict__ S, int E) {
  int idx = blockIdx.x * 256 + threadIdx.x;
  if (idx >= E * 32) return;
  int e = idx >> 5, q = (idx & 31) << 2;
  const float4 v = *(const float4*)(X + (size_t)e * HDIM + q);
  size_t sb = (size_t)src[e] * HDIM + q;
  size_t db = (size_t)dst[e] * HDIM + q;
  atomAddF(&S[sb + 0], v.x); atomAddF(&S[sb + 1], v.y);
  atomAddF(&S[sb + 2], v.z); atomAddF(&S[sb + 3], v.w);
  atomAddF(&S[db + 0], v.x); atomAddF(&S[db + 1], v.y);
  atomAddF(&S[db + 2], v.z); atomAddF(&S[db + 3], v.w);
}

// X0[e] = bf16(concat(node[src]+node[dst], S[src]+S[dst]))   [E, 256]
__global__ void build_x0(const float* __restrict__ nr, const float* __restrict__ S,
                         const int* __restrict__ src, const int* __restrict__ dst,
                         bf16* __restrict__ X0, int E) {
  int idx = blockIdx.x * 256 + threadIdx.x;
  if (idx >= E * 32) return;
  int e = idx >> 5, q = (idx & 31) << 2;
  int s = src[e], d = dst[e];
  const float4 a = *(const float4*)(nr + (size_t)s * HDIM + q);
  const float4 b = *(const float4*)(nr + (size_t)d * HDIM + q);
  const float4 c = *(const float4*)(S + (size_t)s * HDIM + q);
  const float4 f = *(const float4*)(S + (size_t)d * HDIM + q);
  union { bf16 b[4]; uint2 u; } pa, pb;
  pa.b[0] = (bf16)(a.x + b.x); pa.b[1] = (bf16)(a.y + b.y);
  pa.b[2] = (bf16)(a.z + b.z); pa.b[3] = (bf16)(a.w + b.w);
  pb.b[0] = (bf16)(c.x + f.x); pb.b[1] = (bf16)(c.y + f.y);
  pb.b[2] = (bf16)(c.z + f.z); pb.b[3] = (bf16)(c.w + f.w);
  *(uint2*)(X0 + (size_t)e * 256 + q)        = pa.u;
  *(uint2*)(X0 + (size_t)e * 256 + 128 + q)  = pb.u;
}

// EN = bf16((1+eps1)*edge_rep + relu(Y*sc + sh))
__global__ void build_edge_new(const bf16* __restrict__ Y,
                               const float* __restrict__ sc,
                               const float* __restrict__ sh,
                               const float* __restrict__ er,
                               const float* __restrict__ eps1,
                               bf16* __restrict__ EN, int E) {
  int idx = blockIdx.x * 256 + threadIdx.x;
  if (idx >= E * 32) return;
  int e = idx >> 5, q = (idx & 31) << 2;
  float s1 = 1.f + eps1[0];
  union { bf16 b[4]; uint2 u; } yv, o;
  yv.u = *(const uint2*)(Y + (size_t)e * HDIM + q);
  const float4 r = *(const float4*)(er + (size_t)e * HDIM + q);
  const float rr[4] = {r.x, r.y, r.z, r.w};
#pragma unroll
  for (int i = 0; i < 4; ++i) {
    int c = q + i;
    float v = fmaxf((float)yv.b[i] * sc[c] + sh[c], 0.f);
    o.b[i] = (bf16)(s1 * rr[i] + v);
  }
  *(uint2*)(EN + (size_t)e * HDIM + q) = o.u;
}

// edge2node scatter with fused BN+ReLU on Y2
__global__ void scatter_e2n(const bf16* __restrict__ Y,
                            const float* __restrict__ sc,
                            const float* __restrict__ sh,
                            const int* __restrict__ src,
                            const int* __restrict__ dst,
                            float* __restrict__ e2n, int E) {
  int idx = blockIdx.x * 256 + threadIdx.x;
  if (idx >= E * 32) return;
  int e = idx >> 5, q = (idx & 31) << 2;
  union { bf16 b[4]; uint2 u; } yv;
  yv.u = *(const uint2*)(Y + (size_t)e * HDIM + q);
  size_t sb = (size_t)src[e] * HDIM + q;
  size_t db = (size_t)dst[e] * HDIM + q;
#pragma unroll
  for (int i = 0; i < 4; ++i) {
    int c = q + i;
    float v = fmaxf((float)yv.b[i] * sc[c] + sh[c], 0.f);
    atomAddF(&e2n[sb + i], v);
    atomAddF(&e2n[db + i], v);
  }
}

// NI = bf16((1+eps2-deg)*node_rep + edge2node)
__global__ void build_node_in(const float* __restrict__ nr,
                              const float* __restrict__ deg,
                              const float* __restrict__ e2n,
                              const float* __restrict__ eps2,
                              bf16* __restrict__ NI, int N) {
  int idx = blockIdx.x * 256 + threadIdx.x;
  if (idx >= N * 32) return;
  int n = idx >> 5, q = (idx & 31) << 2;
  float w = 1.f + eps2[0] - deg[n];
  const float4 a = *(const float4*)(nr + (size_t)n * HDIM + q);
  const float4 b = *(const float4*)(e2n + (size_t)n * HDIM + q);
  union { bf16 b[4]; uint2 u; } o;
  o.b[0] = (bf16)(w * a.x + b.x); o.b[1] = (bf16)(w * a.y + b.y);
  o.b[2] = (bf16)(w * a.z + b.z); o.b[3] = (bf16)(w * a.w + b.w);
  *(uint2*)(NI + (size_t)n * HDIM + q) = o.u;
}

// x = relu(x*sc + sh) in place on f32 [rows, 128]
__global__ void bnrelu_inplace(float* __restrict__ X, const float* __restrict__ sc,
                               const float* __restrict__ sh, int rows) {
  int idx = blockIdx.x * 256 + threadIdx.x;
  if (idx >= rows * 32) return;
  int r = idx >> 5, q = (idx & 31) << 2;
  float4 x = *(const float4*)(X + (size_t)r * HDIM + q);
  x.x = fmaxf(x.x * sc[q + 0] + sh[q + 0], 0.f);
  x.y = fmaxf(x.y * sc[q + 1] + sh[q + 1], 0.f);
  x.z = fmaxf(x.z * sc[q + 2] + sh[q + 2], 0.f);
  x.w = fmaxf(x.w * sc[q + 3] + sh[q + 3], 0.f);
  *(float4*)(X + (size_t)r * HDIM + q) = x;
}

// ---------------------------------------------------------------------------
static inline int cdiv(int a, int b) { return (a + b - 1) / b; }

extern "C" void kernel_launch(void* const* d_in, const int* in_sizes, int n_in,
                              void* d_out, int out_size, void* d_ws, size_t ws_size,
                              hipStream_t stream) {
  const float* node_rep = (const float*)d_in[0];
  const float* edge_rep = (const float*)d_in[1];
  const float* degree   = (const float*)d_in[2];
  const int*   src      = (const int*)d_in[3];
  const int*   dst      = (const int*)d_in[4];
  const float* eps1     = (const float*)d_in[5];
  const float* eps2     = (const float*)d_in[6];
  // MLP params, dict order: n, e0, e1, e2 ; each W1,G1,B1,W2,G2,B2
  const float* P[24];
  for (int i = 0; i < 24; ++i) P[i] = (const float*)d_in[7 + i];
  const float *nW1 = P[0],  *nG1 = P[1],  *nB1 = P[2],  *nW2 = P[3],  *nG2 = P[4],  *nB2 = P[5];
  const float *e0W1 = P[6], *e0G1 = P[7], *e0B1 = P[8], *e0W2 = P[9], *e0G2 = P[10], *e0B2 = P[11];
  const float *e1W1 = P[12],*e1G1 = P[13],*e1B1 = P[14],*e1W2 = P[15],*e1G2 = P[16],*e1B2 = P[17];
  const float *e2W1 = P[18],*e2G1 = P[19],*e2B1 = P[20],*e2W2 = P[21],*e2G2 = P[22],*e2B2 = P[23];

  const int N = in_sizes[0] / HDIM;
  const int E = in_sizes[1] / HDIM;

  // ---- workspace carve (256B aligned) ----
  uintptr_t p = ((uintptr_t)d_ws + 255) & ~(uintptr_t)255;
  auto alloc = [&](size_t bytes) -> void* {
    void* r = (void*)p;
    p += (bytes + 255) & ~(size_t)255;
    return r;
  };
  float* S   = (float*)alloc((size_t)N * HDIM * 4);      // segsum / edge2node (reused)
  bf16* X0   = (bf16*)alloc((size_t)E * 256 * 2);        // concat input, bf16
  bf16* H1   = (bf16*)alloc((size_t)E * 256 * 2);        // hidden (pre-BN) bf16
  bf16* Ybuf = (bf16*)alloc((size_t)E * HDIM * 2);       // MLP out (pre-BN) bf16
  bf16* EN   = (bf16*)alloc((size_t)E * HDIM * 2);       // edge_new bf16
  bf16* NI   = (bf16*)alloc((size_t)N * HDIM * 2);       // node_in bf16
  bf16* pkE0W1 = (bf16*)alloc(256 * 256 * 2);
  bf16* pkE0W2 = (bf16*)alloc(256 * 128 * 2);
  bf16* pkE1W1 = (bf16*)alloc(128 * 256 * 2);
  bf16* pkE1W2 = (bf16*)alloc(256 * 128 * 2);
  bf16* pkE2W1 = (bf16*)alloc(128 * 256 * 2);
  bf16* pkE2W2 = (bf16*)alloc(256 * 128 * 2);
  bf16* pkNW1  = (bf16*)alloc(128 * 256 * 2);
  bf16* pkNW2  = (bf16*)alloc(256 * 128 * 2);
  float* stats = (float*)alloc(8 * 2 * 256 * 4);  // per stage: sum[256], sq[256]
  float* bnab  = (float*)alloc(8 * 2 * 256 * 4);  // per stage: sc[256], sh[256]
  auto SUM = [&](int s) { return stats + (size_t)s * 512; };
  auto SQ  = [&](int s) { return stats + (size_t)s * 512 + 256; };
  auto SC  = [&](int s) { return bnab  + (size_t)s * 512; };
  auto SH  = [&](int s) { return bnab  + (size_t)s * 512 + 256; };

  hipMemsetAsync(stats, 0, 8 * 2 * 256 * 4, stream);
  hipMemsetAsync(S, 0, (size_t)N * HDIM * 4, stream);

  // ---- weight packing (bf16, WMMA B-fragment order) ----
  pack_w<<<cdiv(256 * 256, 256), 256, 0, stream>>>(e0W1, pkE0W1, 256, 256);
  pack_w<<<cdiv(256 * 128, 256), 256, 0, stream>>>(e0W2, pkE0W2, 256, 128);
  pack_w<<<cdiv(128 * 256, 256), 256, 0, stream>>>(e1W1, pkE1W1, 128, 256);
  pack_w<<<cdiv(256 * 128, 256), 256, 0, stream>>>(e1W2, pkE1W2, 256, 128);
  pack_w<<<cdiv(128 * 256, 256), 256, 0, stream>>>(e2W1, pkE2W1, 128, 256);
  pack_w<<<cdiv(256 * 128, 256), 256, 0, stream>>>(e2W2, pkE2W2, 256, 128);
  pack_w<<<cdiv(128 * 256, 256), 256, 0, stream>>>(nW1, pkNW1, 128, 256);
  pack_w<<<cdiv(256 * 128, 256), 256, 0, stream>>>(nW2, pkNW2, 256, 128);

  const int eBlk = cdiv(E * 32, 256);
  const int nBlk = cdiv(N * 32, 256);
  const int gxE = cdiv(E, 128), gxN = cdiv(N, 128);
  const float invE = 1.f / (float)E, invN = 1.f / (float)N;

  // ---- gathers ----
  scatter_sum2<<<eBlk, 256, 0, stream>>>(edge_rep, src, dst, S, E);
  build_x0<<<eBlk, 256, 0, stream>>>(node_rep, S, src, dst, X0, E);

  // ---- e0 MLP: [E,256] -> 256 -> 128 ----
  gemm_bn<true><<<dim3(gxE, 2), 256, 0, stream>>>(X0, E, 256, pkE0W1, 256,
                                                  nullptr, nullptr, H1, SUM(0), SQ(0));
  bn_finalize<<<1, 256, 0, stream>>>(SUM(0), SQ(0), e0G1, e0B1, invE, 256, SC(0), SH(0));
  gemm_bn<true><<<dim3(gxE, 1), 256, 0, stream>>>(H1, E, 256, pkE0W2, 128,
                                                  SC(0), SH(0), Ybuf, SUM(1), SQ(1));
  bn_finalize<<<1, 256, 0, stream>>>(SUM(1), SQ(1), e0G2, e0B2, invE, 128, SC(1), SH(1));
  build_edge_new<<<eBlk, 256, 0, stream>>>(Ybuf, SC(1), SH(1), edge_rep, eps1, EN, E);

  // ---- e1 MLP -> edge_out (d_out edge section, f32) ----
  float* edge_out = (float*)d_out + (size_t)N * HDIM;
  gemm_bn<true><<<dim3(gxE, 2), 256, 0, stream>>>(EN, E, 128, pkE1W1, 256,
                                                  nullptr, nullptr, H1, SUM(2), SQ(2));
  bn_finalize<<<1, 256, 0, stream>>>(SUM(2), SQ(2), e1G1, e1B1, invE, 256, SC(2), SH(2));
  gemm_bn<false><<<dim3(gxE, 1), 256, 0, stream>>>(H1, E, 256, pkE1W2, 128,
                                                   SC(2), SH(2), edge_out, SUM(3), SQ(3));
  bn_finalize<<<1, 256, 0, stream>>>(SUM(3), SQ(3), e1G2, e1B2, invE, 128, SC(3), SH(3));
  bnrelu_inplace<<<eBlk, 256, 0, stream>>>(edge_out, SC(3), SH(3), E);

  // ---- e2 MLP -> edge2node messages ----
  gemm_bn<true><<<dim3(gxE, 2), 256, 0, stream>>>(EN, E, 128, pkE2W1, 256,
                                                  nullptr, nullptr, H1, SUM(4), SQ(4));
  bn_finalize<<<1, 256, 0, stream>>>(SUM(4), SQ(4), e2G1, e2B1, invE, 256, SC(4), SH(4));
  gemm_bn<true><<<dim3(gxE, 1), 256, 0, stream>>>(H1, E, 256, pkE2W2, 128,
                                                  SC(4), SH(4), Ybuf, SUM(5), SQ(5));
  bn_finalize<<<1, 256, 0, stream>>>(SUM(5), SQ(5), e2G2, e2B2, invE, 128, SC(5), SH(5));

  hipMemsetAsync(S, 0, (size_t)N * HDIM * 4, stream);  // reuse S as edge2node
  scatter_e2n<<<eBlk, 256, 0, stream>>>(Ybuf, SC(5), SH(5), src, dst, S, E);
  build_node_in<<<nBlk, 256, 0, stream>>>(node_rep, degree, S, eps2, NI, N);

  // ---- n MLP -> node_out (d_out node section) ----
  gemm_bn<true><<<dim3(gxN, 2), 256, 0, stream>>>(NI, N, 128, pkNW1, 256,
                                                  nullptr, nullptr, H1, SUM(6), SQ(6));
  bn_finalize<<<1, 256, 0, stream>>>(SUM(6), SQ(6), nG1, nB1, invN, 256, SC(6), SH(6));
  gemm_bn<false><<<dim3(gxN, 1), 256, 0, stream>>>(H1, N, 256, pkNW2, 128,
                                                   SC(6), SH(6), (float*)d_out, SUM(7), SQ(7));
  bn_finalize<<<1, 256, 0, stream>>>(SUM(7), SQ(7), nG2, nB2, invN, 128, SC(7), SH(7));
  bnrelu_inplace<<<nBlk, 256, 0, stream>>>((float*)d_out, SC(7), SH(7), N);
}
